// CrossAttentionLayer_18296560681680
// MI455X (gfx1250) — matrix operand order repaired and checked by reading, not compiled, not run on hardware
//
#include <hip/hip_runtime.h>
#include <hip/hip_bf16.h>
#include <math.h>

typedef __attribute__((ext_vector_type(16))) _Float16 v16h;
typedef __attribute__((ext_vector_type(8)))  _Float16 v8h;
typedef __attribute__((ext_vector_type(8)))  float    v8f;

#define DEV static __device__ __forceinline__

constexpr int Bsz  = 16;
constexpr int Ntok = 1024;
constexpr int C    = 384;
constexpr int E    = 512;
constexpr int NH   = 4;
constexpr int HD   = 96;      // C / NH
constexpr float EPS_   = 1e-5f;
constexpr float SCALE_ = 0.10206207261596575f; // 1/sqrt(96)

DEV v8f wmma(v16h a, v16h b, v8f c) {
  return __builtin_amdgcn_wmma_f32_16x16x32_f16(false, a, false, b, (short)0, c, false, false);
}

// A-operand (16x32 f16): lane L: M = L&15; element e holds K = e + (L>=16?16:0)
DEV v16h load_a_f16(const _Float16* row, int k0, int hi) {
  v8h c0 = *(const v8h*)(row + k0 + hi * 16);
  v8h c1 = *(const v8h*)(row + k0 + 8 + hi * 16);
  v16h a;
#pragma unroll
  for (int e = 0; e < 8; ++e) { a[e] = c0[e]; a[e + 8] = c1[e]; }
  return a;
}

// Same A layout but source is fp32 x plus a broadcast fp32 move vector
DEV v16h load_a_f32(const float* xrow, const float* mv, int k0, int hi) {
  const float4* p0 = (const float4*)(xrow + k0 + hi * 16);
  const float4* p1 = (const float4*)(xrow + k0 + 8 + hi * 16);
  const float4* q0 = (const float4*)(mv + k0 + hi * 16);
  const float4* q1 = (const float4*)(mv + k0 + 8 + hi * 16);
  float4 x0 = p0[0], x1 = p0[1], y0 = p1[0], y1 = p1[1];
  float4 m0 = q0[0], m1 = q0[1], n0 = q1[0], n1 = q1[1];
  v16h a;
  a[0]  = (_Float16)(x0.x + m0.x); a[1]  = (_Float16)(x0.y + m0.y);
  a[2]  = (_Float16)(x0.z + m0.z); a[3]  = (_Float16)(x0.w + m0.w);
  a[4]  = (_Float16)(x1.x + m1.x); a[5]  = (_Float16)(x1.y + m1.y);
  a[6]  = (_Float16)(x1.z + m1.z); a[7]  = (_Float16)(x1.w + m1.w);
  a[8]  = (_Float16)(y0.x + n0.x); a[9]  = (_Float16)(y0.y + n0.y);
  a[10] = (_Float16)(y0.z + n0.z); a[11] = (_Float16)(y0.w + n0.w);
  a[12] = (_Float16)(y1.x + n1.x); a[13] = (_Float16)(y1.y + n1.y);
  a[14] = (_Float16)(y1.z + n1.z); a[15] = (_Float16)(y1.w + n1.w);
  return a;
}

// ---------------------------------------------------------------------------
// Kernel 1: 14 move vectors  out[vec][b][c] = Linear(SiLU(emb[b]))  (E -> C)
// ---------------------------------------------------------------------------
struct MovePtrs { const float* W[14]; const float* b[14]; };

__global__ __launch_bounds__(256) void movevec_kernel(MovePtrs mp, const float* __restrict__ emb,
                                                      float* __restrict__ out) {
  int idx = blockIdx.x * 256 + threadIdx.x;            // 14*16*384 total, exact grid
  int vec = idx / (Bsz * C);
  int rem = idx % (Bsz * C);
  int b = rem / C, c = rem % C;
  const float* W = mp.W[vec] + (size_t)c * E;
  const float* e = emb + (size_t)b * E;
  float acc = mp.b[vec][c];
  for (int k = 0; k < E; ++k) {
    float s = e[k];
    float sil = s / (1.f + __expf(-s));
    acc = fmaf(W[k], sil, acc);
  }
  out[idx] = acc;
}

// ---------------------------------------------------------------------------
// Kernel 2: cast the three CxC weight matrices to f16 (row-major (out,in))
// ---------------------------------------------------------------------------
struct CastPtrs { const float* src[3]; _Float16* dst[3]; };

__global__ __launch_bounds__(256) void castw_kernel(CastPtrs cp) {
  int idx = blockIdx.x * 256 + threadIdx.x;            // 3*384*384, exact grid
  int which = idx / (C * C);
  int r = idx % (C * C);
  cp.dst[which][r] = (_Float16)cp.src[which][r];
}

// ---------------------------------------------------------------------------
// Kernel 3: fused  Y = BN( (X+move1) @ W^T + bias ) + X  -> RPReLU -> outputs
//   sumbuf (fp32): q (+= for k,v)  |  outh (f16): Y_rprelu + move2
//   transposed=1 stores outh in [b][h][d][n] layout (for V)
//   One wave per 16x64 output tile; 4 waves / block; exact grid (6144 waves)
// ---------------------------------------------------------------------------
__global__ __launch_bounds__(128) void gemm_kernel(
    const float* __restrict__ X, const float* __restrict__ move1,
    const _Float16* __restrict__ Wh, const float* __restrict__ bias,
    const float* __restrict__ gamma, const float* __restrict__ beta,
    const float* __restrict__ rmean, const float* __restrict__ rvar,
    const float* __restrict__ m1, const float* __restrict__ alpha,
    const float* __restrict__ m2, const float* __restrict__ move2,
    float* __restrict__ sumbuf, int accumulate,
    _Float16* __restrict__ outh, int transposed) {
  int wid  = blockIdx.x * 4 + (threadIdx.x >> 5);
  int lane = threadIdx.x & 31;
  int nt = wid % 6, mt = wid / 6;          // 6 N-macrotiles of 64, 1024 M-tiles of 16
  int m0 = mt * 16, n0 = nt * 64;
  int b  = m0 >> 10;                       // 16-row tile never straddles a batch
  int col = lane & 15, hi = lane >> 4;

  const float* xrow = X + (size_t)(m0 + col) * C;
  const float* mv   = move1 + (size_t)b * C;

  v8f acc[4] = {};
  for (int ks = 0; ks < 12; ++ks) {
    int k0 = ks * 32;
    v16h a = load_a_f32(xrow, mv, k0, hi);
    const _Float16* wb = Wh + (size_t)(n0 + col) * C + k0 + hi * 16;
    v16h b0 = *(const v16h*)(wb);
    v16h b1 = *(const v16h*)(wb + 16 * C);
    v16h b2 = *(const v16h*)(wb + 32 * C);
    v16h b3 = *(const v16h*)(wb + 48 * C);
    acc[0] = wmma(a, b0, acc[0]);
    acc[1] = wmma(a, b1, acc[1]);
    acc[2] = wmma(a, b2, acc[2]);
    acc[3] = wmma(a, b3, acc[3]);
  }

#pragma unroll
  for (int t = 0; t < 4; ++t) {
    int o = n0 + 16 * t + col;
    float inv = gamma[o] * rsqrtf(rvar[o] + EPS_);
    float bo = beta[o], rm = rmean[o], bs = bias[o];
    float m1v = m1[(size_t)b * C + o], m2v = m2[(size_t)b * C + o];
    float al = alpha[o], mv2 = move2[(size_t)b * C + o];
    int hloc = o / HD, dloc = o % HD;
#pragma unroll
    for (int r = 0; r < 8; ++r) {
      int mrow = m0 + r + hi * 8;
      size_t si = (size_t)mrow * C + o;
      float val = acc[t][r] + bs;            // linear
      val = (val - rm) * inv + bo;           // BN (eval)
      val += X[si];                          // residual
      float z  = val + m1v;                  // RPReLU
      float pr = (z >= 0.f) ? z : al * z;
      float v2 = pr + m2v;
      if (accumulate) sumbuf[si] += v2; else sumbuf[si] = v2;
      _Float16 hv = (_Float16)(v2 + mv2);    // + move2 -> attention operand
      if (!transposed) {
        outh[si] = hv;
      } else {
        int ntk = mrow & (Ntok - 1);
        outh[(size_t)((b * NH + hloc) * HD + dloc) * Ntok + ntk] = hv;
      }
    }
  }
}

// ---------------------------------------------------------------------------
// Kernel 4: flash attention + fused BN/residual/RPReLU epilogue.
//   One wave = 16 queries of one (b,h); 8 waves/block; exact grid (4096 waves)
// ---------------------------------------------------------------------------
__global__ __launch_bounds__(256) void attn_kernel(
    const _Float16* __restrict__ ql, const _Float16* __restrict__ kl,
    const _Float16* __restrict__ vlT, const float* __restrict__ qkvsum,
    const float* __restrict__ gamma, const float* __restrict__ beta,
    const float* __restrict__ rmean, const float* __restrict__ rvar,
    const float* __restrict__ m1c, const float* __restrict__ alphac,
    const float* __restrict__ m2c, float* __restrict__ out) {
  __shared__ _Float16 pbuf[8 * 16 * 32];            // per-wave P staging (1KB each)
  int widx = threadIdx.x >> 5;
  int lane = threadIdx.x & 31;
  int w  = blockIdx.x * 8 + widx;
  int qt = w & 63;
  int h  = (w >> 6) & 3;
  int b  = w >> 8;
  int q0 = qt * 16;
  int col = lane & 15, hi = lane >> 4;
  _Float16* pb = pbuf + widx * 512;

  const _Float16* qrow  = ql + (size_t)(b * Ntok + q0 + col) * C + h * HD;
  const _Float16* kbase = kl + (size_t)(b * Ntok) * C + h * HD;
  const _Float16* vbase = vlT + (size_t)((b * NH + h) * HD) * Ntok;

  v16h aq[3];
#pragma unroll
  for (int c = 0; c < 3; ++c) aq[c] = load_a_f16(qrow, c * 32, hi);

  v8f acc[6] = {};
  float mrun[8], lrun[8];
#pragma unroll
  for (int r = 0; r < 8; ++r) { mrun[r] = -INFINITY; lrun[r] = 0.f; }

  for (int kb = 0; kb < Ntok; kb += 32) {
    // S = scale * Q K^T for 32 keys (two 16x16 tiles, K-dim 96 in 3 steps)
    v8f s0 = {}, s1 = {};
#pragma unroll
    for (int c = 0; c < 3; ++c) {
      v16h bk0 = *(const v16h*)(kbase + (size_t)(kb + col) * C + c * 32 + hi * 16);
      v16h bk1 = *(const v16h*)(kbase + (size_t)(kb + 16 + col) * C + c * 32 + hi * 16);
      s0 = wmma(aq[c], bk0, s0);
      s1 = wmma(aq[c], bk1, s1);
    }
    // online softmax: per-row stats live in the 16-lane half-group owning the row
    float p0[8], p1[8], fac[8];
#pragma unroll
    for (int r = 0; r < 8; ++r) {
      float x0 = s0[r] * SCALE_, x1 = s1[r] * SCALE_;
      float rm = fmaxf(x0, x1);
      rm = fmaxf(rm, __shfl_xor(rm, 1, 32));
      rm = fmaxf(rm, __shfl_xor(rm, 2, 32));
      rm = fmaxf(rm, __shfl_xor(rm, 4, 32));
      rm = fmaxf(rm, __shfl_xor(rm, 8, 32));
      float mnew = fmaxf(mrun[r], rm);
      float f = __expf(mrun[r] - mnew);
      mrun[r] = mnew; fac[r] = f; lrun[r] *= f;
      p0[r] = __expf(x0 - mnew);
      p1[r] = __expf(x1 - mnew);
      float rs = p0[r] + p1[r];
      rs += __shfl_xor(rs, 1, 32);
      rs += __shfl_xor(rs, 2, 32);
      rs += __shfl_xor(rs, 4, 32);
      rs += __shfl_xor(rs, 8, 32);
      lrun[r] += rs;
    }
#pragma unroll
    for (int t = 0; t < 6; ++t)
#pragma unroll
      for (int r = 0; r < 8; ++r) acc[t][r] *= fac[r];

    // C-layout -> A-layout for P via wave-private LDS slice (in-order DS ops)
#pragma unroll
    for (int r = 0; r < 8; ++r) {
      pb[(r + hi * 8) * 32 + col]      = (_Float16)p0[r];
      pb[(r + hi * 8) * 32 + 16 + col] = (_Float16)p1[r];
    }
    v16h ap;
    {
      v8h c0 = *(const v8h*)(pb + col * 32 + hi * 16);
      v8h c1 = *(const v8h*)(pb + col * 32 + 8 + hi * 16);
#pragma unroll
      for (int e = 0; e < 8; ++e) { ap[e] = c0[e]; ap[e + 8] = c1[e]; }
    }
    // ctx += P @ V  (V^T layout makes B-operand loads contiguous)
#pragma unroll
    for (int t = 0; t < 6; ++t) {
      v16h bv = *(const v16h*)(vbase + (size_t)(16 * t + col) * Ntok + kb + hi * 16);
      acc[t] = wmma(ap, bv, acc[t]);
    }
  }

  // epilogue: ctx/l -> BN -> + (q+k+v) -> RPReLU -> out
#pragma unroll
  for (int t = 0; t < 6; ++t) {
    int o = h * HD + 16 * t + col;
    float inv = gamma[o] * rsqrtf(rvar[o] + EPS_);
    float bo = beta[o], rm = rmean[o];
    float m1v = m1c[(size_t)b * C + o], m2v = m2c[(size_t)b * C + o];
    float al = alphac[o];
#pragma unroll
    for (int r = 0; r < 8; ++r) {
      size_t mrow = (size_t)(b * Ntok + q0 + r + hi * 8);
      float cv = acc[t][r] / lrun[r];
      float y  = (cv - rm) * inv + bo + qkvsum[mrow * C + o];
      float z  = y + m1v;
      float pr = (z >= 0.f) ? z : al * z;
      out[mrow * C + o] = pr + m2v;
    }
  }
}

// ---------------------------------------------------------------------------
extern "C" void kernel_launch(void* const* d_in, const int* in_sizes, int n_in,
                              void* d_out, int out_size, void* d_ws, size_t ws_size,
                              hipStream_t stream) {
  (void)in_sizes; (void)n_in; (void)out_size; (void)ws_size;
  const float* cond = (const float*)d_in[0];
  const float* embx = (const float*)d_in[1];
  const float* emb  = (const float*)d_in[2];
  auto P = [&](int i) { return (const float*)d_in[i]; };

  auto alup = [](size_t x) { return (x + 255) & ~(size_t)255; };
  char* wsb = (char*)d_ws;
  size_t off = 0;
  float* moves = (float*)(wsb + off); off += alup((size_t)14 * Bsz * C * 4);
  _Float16* Whq = (_Float16*)(wsb + off); off += alup((size_t)C * C * 2);
  _Float16* Whk = (_Float16*)(wsb + off); off += alup((size_t)C * C * 2);
  _Float16* Whv = (_Float16*)(wsb + off); off += alup((size_t)C * C * 2);
  _Float16* qlh = (_Float16*)(wsb + off); off += alup((size_t)Bsz * Ntok * C * 2);
  _Float16* klh = (_Float16*)(wsb + off); off += alup((size_t)Bsz * Ntok * C * 2);
  _Float16* vlT = (_Float16*)(wsb + off); off += alup((size_t)Bsz * Ntok * C * 2);
  float* qkvsum = (float*)(wsb + off);    off += alup((size_t)Bsz * Ntok * C * 4);

  // move vectors: 0 mq1, 1 mk1, 2 mv1, 3 q.m1, 4 q.m2, 5 k.m1, 6 k.m2,
  //               7 v.m1, 8 v.m2, 9 mq2, 10 mk2, 11 mv2, 12 ctx.m1, 13 ctx.m2
  MovePtrs mp;
  const int wi[14] = {3, 5, 7, 27, 30, 32, 35, 37, 40, 42, 44, 46, 52, 55};
  for (int i = 0; i < 14; ++i) { mp.W[i] = P(wi[i]); mp.b[i] = P(wi[i] + 1); }
  auto MV = [&](int i) { return moves + (size_t)i * Bsz * C; };

  movevec_kernel<<<(14 * Bsz * C) / 256, 256, 0, stream>>>(mp, emb, moves);

  CastPtrs cp;
  cp.src[0] = P(9); cp.src[1] = P(11); cp.src[2] = P(13);
  cp.dst[0] = Whq;  cp.dst[1] = Whk;   cp.dst[2] = Whv;
  castw_kernel<<<(3 * C * C) / 256, 256, 0, stream>>>(cp);

  const int gemm_blocks = (Bsz * Ntok / 16) * (C / 64) / 4;  // 1536
  // Q: X = embedding_embedding
  gemm_kernel<<<gemm_blocks, 128, 0, stream>>>(
      embx, MV(0), Whq, P(10), P(15), P(16), P(17), P(18),
      MV(3), P(29), MV(4), MV(9), qkvsum, 0, qlh, 0);
  // K: X = condition_embedding
  gemm_kernel<<<gemm_blocks, 128, 0, stream>>>(
      cond, MV(1), Whk, P(12), P(19), P(20), P(21), P(22),
      MV(5), P(34), MV(6), MV(10), qkvsum, 1, klh, 0);
  // V: X = condition_embedding, transposed f16 output
  gemm_kernel<<<gemm_blocks, 128, 0, stream>>>(
      cond, MV(2), Whv, P(14), P(23), P(24), P(25), P(26),
      MV(7), P(39), MV(8), MV(11), qkvsum, 1, vlT, 1);

  const int attn_blocks = (Bsz * NH * (Ntok / 16)) / 8;      // 512
  attn_kernel<<<attn_blocks, 256, 0, stream>>>(
      qlh, klh, vlT, qkvsum, P(48), P(49), P(50), P(51),
      MV(12), P(54), MV(13), (float*)d_out);
}
